// Encoder_57947698757863
// MI455X (gfx1250) — matrix operand order repaired and checked by reading, not compiled, not run on hardware
//
#include <hip/hip_runtime.h>

#define B_   4
#define S_   2048
#define D_   512
#define DFF_ 2048
#define BS_  (B_ * S_)

typedef unsigned short u16;
typedef unsigned int   u32;
typedef __attribute__((ext_vector_type(16))) __bf16 v16bf;
typedef __attribute__((ext_vector_type(8)))  float  v8f;
typedef __attribute__((ext_vector_type(4)))  u32    v4u;
typedef __attribute__((ext_vector_type(4)))  int    v4i;

#if defined(__HIP_DEVICE_COMPILE__) && __has_builtin(__builtin_amdgcn_global_load_async_to_lds_b128)
#define HAVE_ASYNC_LDS 1
#else
#define HAVE_ASYNC_LDS 0
#endif

__device__ __forceinline__ u16 f2bf(float f) {
    u32 u = __float_as_uint(f);
    u32 r = u + 0x7FFFu + ((u >> 16) & 1u);   // round-to-nearest-even
    return (u16)(r >> 16);
}
__device__ __forceinline__ float bf2f(u16 h) {
    return __uint_as_float(((u32)h) << 16);
}
// 16 bf16 elements from two contiguous 16B chunks (ds_load_b128 x2)
__device__ __forceinline__ v16bf load_frag(const u16* p0, const u16* p1) {
    union { v16bf bf; v4u q[2]; } u;
    u.q[0] = *(const v4u*)p0;
    u.q[1] = *(const v4u*)p1;
    return u.bf;
}

__device__ __forceinline__ void async_wait0() {
#if HAVE_ASYNC_LDS
#if __has_builtin(__builtin_amdgcn_s_wait_asynccnt)
    __builtin_amdgcn_s_wait_asynccnt(0);
#else
    asm volatile("s_wait_asynccnt 0x0" ::: "memory");
#endif
#endif
}

// One 16B global -> LDS transfer (async on CDNA5, sync fallback otherwise)
__device__ __forceinline__ void copy16B_to_lds(const u16* gsrc, u16* ldst) {
#if HAVE_ASYNC_LDS
    typedef __attribute__((address_space(1))) v4i gv4i;   // prints as "__device__" in diags
    typedef __attribute__((address_space(3))) v4i lv4i;   // LDS
    __builtin_amdgcn_global_load_async_to_lds_b128((gv4i*)gsrc, (lv4i*)ldst, 0, 0);
#else
    *(v4u*)ldst = *(const v4u*)gsrc;
#endif
}

// ---------------------------------------------------------------------------
// Weight convert + transpose: w[K][N] f32 -> wt[N][K] bf16
// ---------------------------------------------------------------------------
__global__ void conv_transpose_bf16(const float* __restrict__ w, u16* __restrict__ wt,
                                    int K, int N) {
    size_t idx = (size_t)blockIdx.x * blockDim.x + threadIdx.x;
    if (idx >= (size_t)K * N) return;
    int k = (int)(idx / N), n = (int)(idx % N);
    wt[(size_t)n * K + k] = f2bf(w[idx]);
}

// ---------------------------------------------------------------------------
// x = (emb[seq] + pos) * sqrt(D): f32 + bf16 copies
// ---------------------------------------------------------------------------
__global__ void embed_kernel(const int* __restrict__ seq, const float* __restrict__ emb,
                             float* __restrict__ xf, u16* __restrict__ xb) {
    size_t idx = (size_t)blockIdx.x * blockDim.x + threadIdx.x;
    if (idx >= (size_t)BS_ * D_) return;
    int i   = (int)(idx % D_);
    int row = (int)(idx / D_);
    int s   = row % S_;
    int tok = seq[row];
    float e   = emb[(size_t)tok * D_ + i];
    float ang = (float)s * __expf(-(2.0f * (float)i / (float)D_) * 9.210340371976184f);
    float p   = ((i & 1) == 0) ? __sinf(ang) : __cosf(ang);
    float x   = (e + p) * 22.62741699796952f;   // sqrt(512)
    xf[idx] = x;
    xb[idx] = f2bf(x);
}

// ---------------------------------------------------------------------------
// Generic batched bf16 WMMA GEMM:
//   C[m][n] = act( scale * sum_k A[m][k]*Bt[n][k] + bias[n] )
// Block tile 128x128, 8 waves (2Mx4N) each 64x32 (4x2 WMMA frags), K step 32.
// Double-buffered LDS fed by GLOBAL_LOAD_ASYNC_TO_LDS_B128 (ASYNCcnt).
// Store modes: Cf (f32), Cb (bf16), CbT (bf16 transposed per t_rows chunk).
// ---------------------------------------------------------------------------
#define GBM 128
#define GBN 128
#define GBK 32
#define LDP 40   // padded LDS row (ushorts); 80B keeps 16B alignment, skews banks

__global__ __launch_bounds__(256)
void gemm_bf16(const u16* __restrict__ A, long long sA,
               const u16* __restrict__ Bt, long long sB,
               float* Cf, u16* Cb, u16* CbT, long long sC,
               int t_rows, const float* __restrict__ bias,
               int N, int K, float scale, int act) {
    __shared__ u16 As[2][GBM][LDP];
    __shared__ u16 Bs[2][GBN][LDP];

    const int tid  = threadIdx.x;
    const int lane = tid & 31;
    const int wave = tid >> 5;
    const int bz   = blockIdx.z;
    const int m0   = blockIdx.y * GBM;
    const int n0   = blockIdx.x * GBN;

    const u16* Ab = A  + (size_t)bz * (size_t)sA;
    const u16* Bb = Bt + (size_t)bz * (size_t)sB;

    const int wm0 = (wave & 1) * 64;   // wave M origin in tile
    const int wn0 = (wave >> 1) * 32;  // wave N origin in tile

    v8f zero = {0.f,0.f,0.f,0.f,0.f,0.f,0.f,0.f};
    v8f acc[4][2];
#pragma unroll
    for (int i = 0; i < 4; i++)
#pragma unroll
        for (int j = 0; j < 2; j++) acc[i][j] = zero;

    const int srow = tid >> 1, shalf = tid & 1;  // staging: 16 ushorts/thread/tile
    const int mh = lane >> 4, ml = lane & 15;

    // issue one K-stage (A+B tiles) into buffer `buf`
    auto issue_stage = [&](int buf, int kk) {
        const u16* asrc = Ab + (size_t)(m0 + srow) * K + kk + shalf * 16;
        const u16* bsrc = Bb + (size_t)(n0 + srow) * K + kk + shalf * 16;
        u16* adst = &As[buf][srow][shalf * 16];
        u16* bdst = &Bs[buf][srow][shalf * 16];
        copy16B_to_lds(asrc,     adst);
        copy16B_to_lds(asrc + 8, adst + 8);
        copy16B_to_lds(bsrc,     bdst);
        copy16B_to_lds(bsrc + 8, bdst + 8);
    };

    issue_stage(0, 0);
    int cur = 0;

    for (int kk = 0; kk < K; kk += GBK) {
        async_wait0();        // my async fills for `cur` landed
        __syncthreads();      // everyone's landed; everyone done with cur^1
        if (kk + GBK < K) issue_stage(cur ^ 1, kk + GBK);

        // A frag (16x32 bf16): lanes 0-15 K 0-7/16-23, lanes 16-31 K 8-15/24-31
        v16bf af[4];
#pragma unroll
        for (int fm = 0; fm < 4; fm++) {
            const u16* r = &As[cur][wm0 + fm * 16 + ml][0];
            af[fm] = load_frag(r + mh * 8, r + 16 + mh * 8);
        }
        // B frag (32x16 bf16): lane n = ml, lanes 0-15 K 0-7/16-23, lanes 16-31 K 8-15/24-31
        v16bf bfr[2];
#pragma unroll
        for (int fn = 0; fn < 2; fn++) {
            const u16* r = &Bs[cur][wn0 + fn * 16 + ml][0];
            bfr[fn] = load_frag(r + mh * 8, r + 16 + mh * 8);
        }

#pragma unroll
        for (int fm = 0; fm < 4; fm++)
#pragma unroll
            for (int fn = 0; fn < 2; fn++)
                acc[fm][fn] = __builtin_amdgcn_wmma_f32_16x16x32_bf16(
                    false, af[fm], false, bfr[fn], (short)0, acc[fm][fn], false, false);
        cur ^= 1;
    }

    // Epilogue: lane holds rows mbase..mbase+7 of column gn
#pragma unroll
    for (int fm = 0; fm < 4; fm++) {
        const int mbase = m0 + wm0 + fm * 16 + mh * 8;
#pragma unroll
        for (int fn = 0; fn < 2; fn++) {
            const int gn  = n0 + wn0 + fn * 16 + ml;
            float bia = bias ? bias[gn] : 0.0f;
            float vals[8];
#pragma unroll
            for (int r = 0; r < 8; r++) {
                float x = acc[fm][fn][r] * scale + bia;
                if (act == 1) x = 0.5f * x * (1.0f + erff(x * 0.7071067811865475f));
                vals[r] = x;
            }
            if (Cf) {
#pragma unroll
                for (int r = 0; r < 8; r++)
                    Cf[(size_t)bz * (size_t)sC + (size_t)(mbase + r) * N + gn] = vals[r];
            }
            if (Cb) {
#pragma unroll
                for (int r = 0; r < 8; r++)
                    Cb[(size_t)bz * (size_t)sC + (size_t)(mbase + r) * N + gn] = f2bf(vals[r]);
            }
            if (CbT) {   // CbT[tb][n][mloc], rows contiguous -> one 16B store
                union { v4u q; u16 s[8]; } o;
#pragma unroll
                for (int r = 0; r < 8; r++) o.s[r] = f2bf(vals[r]);
                int tb = mbase / t_rows, mloc = mbase % t_rows;
                *(v4u*)&CbT[((size_t)tb * N + gn) * t_rows + mloc] = o.q;
            }
        }
    }
}

// ---------------------------------------------------------------------------
// In-place softmax over rows of bf16 scores (row length S_, 256 thr, 8/thr)
// ---------------------------------------------------------------------------
__global__ __launch_bounds__(256)
void softmax_kernel(u16* __restrict__ p) {
    __shared__ float red[256];
    const int tid = threadIdx.x;
    const size_t base = (size_t)blockIdx.x * S_;
    float v[8];
#pragma unroll
    for (int i = 0; i < 8; i++) v[i] = bf2f(p[base + tid + i * 256]);
    float m = v[0];
#pragma unroll
    for (int i = 1; i < 8; i++) m = fmaxf(m, v[i]);
    red[tid] = m; __syncthreads();
    for (int s = 128; s > 0; s >>= 1) {
        if (tid < s) red[tid] = fmaxf(red[tid], red[tid + s]);
        __syncthreads();
    }
    m = red[0]; __syncthreads();
    float sum = 0.0f;
#pragma unroll
    for (int i = 0; i < 8; i++) { v[i] = __expf(v[i] - m); sum += v[i]; }
    red[tid] = sum; __syncthreads();
    for (int s = 128; s > 0; s >>= 1) {
        if (tid < s) red[tid] += red[tid + s];
        __syncthreads();
    }
    float inv = 1.0f / red[0];
#pragma unroll
    for (int i = 0; i < 8; i++) p[base + tid + i * 256] = f2bf(v[i] * inv);
}

// ---------------------------------------------------------------------------
// y = LayerNorm(a + b) * g + beta ; writes f32 and/or bf16. Row = D_ = 512.
// ---------------------------------------------------------------------------
__global__ __launch_bounds__(256)
void add_ln_kernel(const float* __restrict__ a, const float* __restrict__ b,
                   const float* __restrict__ g, const float* __restrict__ beta,
                   float* __restrict__ yf, u16* __restrict__ yb) {
    __shared__ float red[256];
    const int tid = threadIdx.x;
    const size_t base = (size_t)blockIdx.x * D_;
    float z0 = a[base + tid]       + (b ? b[base + tid]       : 0.0f);
    float z1 = a[base + tid + 256] + (b ? b[base + tid + 256] : 0.0f);
    red[tid] = z0 + z1; __syncthreads();
    for (int s = 128; s > 0; s >>= 1) {
        if (tid < s) red[tid] += red[tid + s];
        __syncthreads();
    }
    float mu = red[0] * (1.0f / D_); __syncthreads();
    float d0 = z0 - mu, d1 = z1 - mu;
    red[tid] = d0 * d0 + d1 * d1; __syncthreads();
    for (int s = 128; s > 0; s >>= 1) {
        if (tid < s) red[tid] += red[tid + s];
        __syncthreads();
    }
    float rs = rsqrtf(red[0] * (1.0f / D_) + 1e-5f);
    float y0 = d0 * rs * g[tid]       + beta[tid];
    float y1 = d1 * rs * g[tid + 256] + beta[tid + 256];
    if (yf) { yf[base + tid] = y0; yf[base + tid + 256] = y1; }
    if (yb) { yb[base + tid] = f2bf(y0); yb[base + tid + 256] = f2bf(y1); }
}

// ---------------------------------------------------------------------------
extern "C" void kernel_launch(void* const* d_in, const int* in_sizes, int n_in,
                              void* d_out, int out_size, void* d_ws, size_t ws_size,
                              hipStream_t stream) {
    const int*   seq  = (const int*)  d_in[0];
    const float* emb  = (const float*)d_in[1];
    const float* wk   = (const float*)d_in[2];
    const float* bk   = (const float*)d_in[3];
    const float* wq   = (const float*)d_in[4];
    const float* bq   = (const float*)d_in[5];
    const float* wv   = (const float*)d_in[6];
    const float* bv   = (const float*)d_in[7];
    const float* ln_g = (const float*)d_in[8];
    const float* ln_b = (const float*)d_in[9];
    const float* w1   = (const float*)d_in[10];
    const float* b1   = (const float*)d_in[11];
    const float* w2   = (const float*)d_in[12];
    const float* b2   = (const float*)d_in[13];
    float* out = (float*)d_out;

    char* base = (char*)d_ws;
    size_t off = 0;
    auto alloc = [&](size_t bytes) -> void* {
        void* p = base + off;
        off = (off + bytes + 255) & ~(size_t)255;
        return p;
    };
    u16*   wkT    = (u16*)  alloc((size_t)D_ * D_ * 2);
    u16*   wqT    = (u16*)  alloc((size_t)D_ * D_ * 2);
    u16*   wvT    = (u16*)  alloc((size_t)D_ * D_ * 2);
    u16*   w1T    = (u16*)  alloc((size_t)D_ * DFF_ * 2);
    u16*   w2T    = (u16*)  alloc((size_t)DFF_ * D_ * 2);
    float* xf     = (float*)alloc((size_t)BS_ * D_ * 4);
    u16*   xb     = (u16*)  alloc((size_t)BS_ * D_ * 2);
    u16*   kb     = (u16*)  alloc((size_t)BS_ * D_ * 2);
    u16*   qb     = (u16*)  alloc((size_t)BS_ * D_ * 2);
    u16*   vTb    = (u16*)  alloc((size_t)BS_ * D_ * 2);   // [B][D][S]
    u16*   scoreP = (u16*)  alloc((size_t)B_ * S_ * S_ * 2);
    float* attnf  = (float*)alloc((size_t)BS_ * D_ * 4);
    float* rf     = (float*)alloc((size_t)BS_ * D_ * 4);
    u16*   rb     = (u16*)  alloc((size_t)BS_ * D_ * 2);
    u16*   hb     = (u16*)  alloc((size_t)BS_ * DFF_ * 2);
    float* yf     = (float*)alloc((size_t)BS_ * D_ * 4);
    (void)ws_size; (void)in_sizes; (void)n_in; (void)out_size;

    const float inv_sqrt_d = 0.04419417382415922f;  // 1/sqrt(512)

    // 1) weights -> bf16 transposed [N][K]
    conv_transpose_bf16<<<(D_ * D_ + 255) / 256, 256, 0, stream>>>(wk, wkT, D_, D_);
    conv_transpose_bf16<<<(D_ * D_ + 255) / 256, 256, 0, stream>>>(wq, wqT, D_, D_);
    conv_transpose_bf16<<<(D_ * D_ + 255) / 256, 256, 0, stream>>>(wv, wvT, D_, D_);
    conv_transpose_bf16<<<(D_ * DFF_ + 255) / 256, 256, 0, stream>>>(w1, w1T, D_, DFF_);
    conv_transpose_bf16<<<(DFF_ * D_ + 255) / 256, 256, 0, stream>>>(w2, w2T, DFF_, D_);

    // 2) embeddings + positional
    embed_kernel<<<(BS_ * D_ + 255) / 256, 256, 0, stream>>>(seq, emb, xf, xb);

    // 3) QKV projections (note ref swaps: Q:=k_proj, K:=q_proj, V:=v_proj)
    gemm_bf16<<<dim3(D_ / GBN, BS_ / GBM, 1), 256, 0, stream>>>(
        xb, 0, wkT, 0, nullptr, kb, nullptr, 0, S_, bk, D_, D_, 1.0f, 0);
    gemm_bf16<<<dim3(D_ / GBN, BS_ / GBM, 1), 256, 0, stream>>>(
        xb, 0, wqT, 0, nullptr, qb, nullptr, 0, S_, bq, D_, D_, 1.0f, 0);
    gemm_bf16<<<dim3(D_ / GBN, BS_ / GBM, 1), 256, 0, stream>>>(
        xb, 0, wvT, 0, nullptr, nullptr, vTb, 0, S_, bv, D_, D_, 1.0f, 0);

    // 4) scores = k @ q^T / sqrt(D)   (q natural [s][d] layout IS the [N][K] operand)
    gemm_bf16<<<dim3(S_ / GBN, S_ / GBM, B_), 256, 0, stream>>>(
        kb, (long long)S_ * D_, qb, (long long)S_ * D_,
        nullptr, scoreP, nullptr, (long long)S_ * S_, S_, nullptr, S_, D_, inv_sqrt_d, 0);

    // 5) softmax rows in place
    softmax_kernel<<<BS_, 256, 0, stream>>>(scoreP);

    // 6) attn = P @ V  (V^T as [D][S] operand)
    gemm_bf16<<<dim3(D_ / GBN, S_ / GBM, B_), 256, 0, stream>>>(
        scoreP, (long long)S_ * S_, vTb, (long long)D_ * S_,
        attnf, nullptr, nullptr, (long long)S_ * D_, S_, nullptr, D_, S_, 1.0f, 0);

    // 7) r = LN(x + attn)
    add_ln_kernel<<<BS_, 256, 0, stream>>>(xf, attnf, ln_g, ln_b, rf, rb);

    // 8) h = gelu(r @ w1 + b1)
    gemm_bf16<<<dim3(DFF_ / GBN, BS_ / GBM, 1), 256, 0, stream>>>(
        rb, 0, w1T, 0, nullptr, hb, nullptr, 0, S_, b1, DFF_, D_, 1.0f, 1);

    // 9) y = h @ w2 + b2
    gemm_bf16<<<dim3(D_ / GBN, BS_ / GBM, 1), 256, 0, stream>>>(
        hb, 0, w2T, 0, yf, nullptr, nullptr, 0, S_, b2, D_, DFF_, 1.0f, 0);

    // 10) out = LN(r + y)
    add_ln_kernel<<<BS_, 256, 0, stream>>>(rf, yf, ln_g, ln_b, out, nullptr);
}
